// GemmaWithKVCache_5411658793644
// MI455X (gfx1250) — matrix operand order repaired and checked by reading, not compiled, not run on hardware
//
#include <hip/hip_runtime.h>
#include <math.h>
#include <stdint.h>

// ---- model dims ----
#define S_LEN   512
#define HID     2048
#define NHEAD   8
#define KVHEAD  4
#define HDIM    256
#define INTER_DIM 8192
#define VOCAB   32000
#define NLAYER  2
#define ATT_SCALE 0.0625f   // 256^-0.5

// ---- WMMA types (gfx1250, wave32) ----
typedef __bf16 bf16_t;
typedef __bf16 v16bf __attribute__((ext_vector_type(16)));
typedef __bf16 v8bf  __attribute__((ext_vector_type(8)));
typedef float  v8f   __attribute__((ext_vector_type(8)));

__device__ __forceinline__ bf16_t f2bf(float f) {
  unsigned u = __builtin_bit_cast(unsigned, f);
  unsigned r = u + 0x7FFFu + ((u >> 16) & 1u);   // round-to-nearest-even
  unsigned short h = (unsigned short)(r >> 16);
  return __builtin_bit_cast(bf16_t, h);
}

// ---- reductions (wave32) ----
__device__ __forceinline__ float warpReduceSum(float v) {
#pragma unroll
  for (int o = 16; o > 0; o >>= 1) v += __shfl_down(v, o, 32);
  return v;
}
__device__ __forceinline__ float warpReduceMax(float v) {
#pragma unroll
  for (int o = 16; o > 0; o >>= 1) v = fmaxf(v, __shfl_down(v, o, 32));
  return v;
}
template <int NT>
__device__ float blockReduceSum(float v) {
  __shared__ float sm[NT / 32];
  int lane = threadIdx.x & 31, wid = threadIdx.x >> 5;
  v = warpReduceSum(v);
  if (lane == 0) sm[wid] = v;
  __syncthreads();
  v = (threadIdx.x < NT / 32) ? sm[threadIdx.x] : 0.f;
  if (wid == 0) v = warpReduceSum(v);
  if (threadIdx.x == 0) sm[0] = v;
  __syncthreads();
  float r = sm[0];
  __syncthreads();
  return r;
}
template <int NT>
__device__ float blockReduceMax(float v) {
  __shared__ float sm[NT / 32];
  int lane = threadIdx.x & 31, wid = threadIdx.x >> 5;
  v = warpReduceMax(v);
  if (lane == 0) sm[wid] = v;
  __syncthreads();
  v = (threadIdx.x < NT / 32) ? sm[threadIdx.x] : -1e30f;
  if (wid == 0) v = warpReduceMax(v);
  if (threadIdx.x == 0) sm[0] = v;
  __syncthreads();
  float r = sm[0];
  __syncthreads();
  return r;
}

// CDNA5 async global->LDS copy (ASYNCcnt-tracked), GVS addressing:
//   LDS[vdst_lds_addr] = MEM[saddr + voffset]
__device__ __forceinline__ void async_b128(unsigned lds_addr, int voff, const void* base) {
  asm volatile("global_load_async_to_lds_b128 %0, %1, %2"
               :: "v"(lds_addr), "v"(voff), "s"(base) : "memory");
}
__device__ __forceinline__ void wait_asynccnt0() {
  asm volatile("s_wait_asynccnt 0" ::: "memory");
}

// =====================================================================
// Batched/strided GEMM, bf16 WMMA core, software-pipelined + LDS dbuf.
//   A: bf16 [M,K] row-major (lda)           -> async-to-LDS staging
//   bmode 0: B fp32 [K,N] (ldb)             -> load/cvt/ds_store staging
//   bmode 1: B bf16 [N,K] (ldb, transposed) -> async-to-LDS staging
//   cmode 0: C fp32 [M,N]; 1: C bf16 [M,N]; 2: C bf16 [N,M] (transposed)
//   Batch z: A += (z/dA)*sA, B += (z/dB)*sB, C += z*sC (elements).
//   All dims in this network are multiples of BM/BN/BK -> no bounds code.
// =====================================================================
#define BM 128
#define BN 64
#define BK 64
#define AKP (BK + 8)            // padded LDS row (bank-conflict relief)
#define BKP (BK + 8)
#define A_TILE (BM * AKP)
#define B_TILE (BN * BKP)
#define GT 256                  // 8 waves

__global__ __launch_bounds__(GT)
void k_gemm(const bf16_t* __restrict__ A, const void* __restrict__ Bp,
            void* __restrict__ Cp,
            int M, int N, int K, int lda, int ldb, int ldc,
            long long sA, int dA, long long sB, int dB, long long sC,
            int bmode, int cmode) {
  __shared__ __attribute__((aligned(16))) bf16_t As[2 * A_TILE];
  __shared__ __attribute__((aligned(16))) bf16_t Bs[2 * B_TILE];

  const int z = blockIdx.z;
  A += (long long)(z / dA) * sA;
  const bf16_t* Bb = (const bf16_t*)Bp + (long long)(z / dB) * sB;
  const float*  Bf = (const float*)Bp + (long long)(z / dB) * sB;

  const int n0 = blockIdx.x * BN;
  const int m0 = blockIdx.y * BM;
  const int tid = threadIdx.x;
  const int lane = tid & 31;
  const int wv = tid >> 5;              // 8 waves: 4(M) x 2(N), 32x32 each
  const int wm = (wv >> 1) * 32;
  const int wn = (wv & 1) * 32;
  const int lrow = lane & 15;
  const int lhalf = lane >> 4;

  v8f acc[2][2];
  acc[0][0] = {}; acc[0][1] = {}; acc[1][0] = {}; acc[1][1] = {};

  float4 breg[4];                       // in-flight fp32 B tile (bmode 0)

  // ---- staging helpers ----
  auto stage_async = [&](int kt, int bsel) {
    const int k0 = kt * BK;
    // A tile: BM x BK bf16 = 1024 x 16B chunks
#pragma unroll
    for (int it = 0; it < (BM * BK) / (8 * GT); ++it) {
      int idx = tid + it * GT;
      int row = idx >> 3, ck = (idx & 7) * 8;
      int go = (int)(((size_t)(m0 + row) * lda + k0 + ck) * sizeof(bf16_t));
      unsigned lo = (unsigned)(uintptr_t)&As[bsel * A_TILE + row * AKP + ck];
      async_b128(lo, go, A);
    }
    if (bmode == 1) {                   // bf16 [N,K]: raw async copy
#pragma unroll
      for (int it = 0; it < (BN * BK) / (8 * GT); ++it) {
        int idx = tid + it * GT;
        int n = idx >> 3, ck = (idx & 7) * 8;
        int go = (int)(((size_t)(n0 + n) * ldb + k0 + ck) * sizeof(bf16_t));
        unsigned lo = (unsigned)(uintptr_t)&Bs[bsel * B_TILE + n * BKP + ck];
        async_b128(lo, go, Bb);
      }
    }
  };
  auto bf32_load = [&](int kt) {        // issue fp32 weight loads (no wait)
    const int k0 = kt * BK;
#pragma unroll
    for (int it = 0; it < 4; ++it) {
      int idx = tid + it * GT;
      int k = idx >> 4, nq = (idx & 15) * 4;
      breg[it] = *(const float4*)(Bf + (size_t)(k0 + k) * ldb + n0 + nq);
      __builtin_prefetch(Bf + (size_t)(k0 + 2 * BK + k) * ldb + n0 + nq, 0, 0);
    }
  };
  auto bf32_store = [&](int bsel) {     // cvt + transpose into LDS
#pragma unroll
    for (int it = 0; it < 4; ++it) {
      int idx = tid + it * GT;
      int k = idx >> 4, nq = (idx & 15) * 4;
      bf16_t* bp = &Bs[bsel * B_TILE + k];
      bp[(nq + 0) * BKP] = f2bf(breg[it].x);
      bp[(nq + 1) * BKP] = f2bf(breg[it].y);
      bp[(nq + 2) * BKP] = f2bf(breg[it].z);
      bp[(nq + 3) * BKP] = f2bf(breg[it].w);
    }
  };

  // ---- prologue: stage tile 0 into buffer 0 ----
  if (bmode == 0) bf32_load(0);
  stage_async(0, 0);
  if (bmode == 0) bf32_store(0);
  wait_asynccnt0();
  __syncthreads();

  const int nk = K / BK;
  for (int kt = 0; kt < nk; ++kt) {
    const int bsel = kt & 1, nsel = bsel ^ 1;
    const bool more = (kt + 1 < nk);
    if (more) {
      if (bmode == 0) bf32_load(kt + 1);    // global loads in flight ...
      stage_async(kt + 1, nsel);            // ... and async copies in flight
    }
    // ---- WMMA on buffer bsel: 2 k-steps x 2x2 tiles = 8 wmma ----
#pragma unroll
    for (int ks = 0; ks < 2; ++ks) {
      const bf16_t* as0 = &As[bsel * A_TILE + ks * 32];
      const bf16_t* bs0 = &Bs[bsel * B_TILE + ks * 32];
#pragma unroll
      for (int mi = 0; mi < 2; ++mi) {
        const bf16_t* ap = as0 + (wm + mi * 16 + lrow) * AKP;
        v8bf alo = *(const v8bf*)(ap + 8 * lhalf);
        v8bf ahi = *(const v8bf*)(ap + 16 + 8 * lhalf);
        v16bf af;
#pragma unroll
        for (int e = 0; e < 8; ++e) { af[e] = alo[e]; af[e + 8] = ahi[e]; }
#pragma unroll
        for (int ni = 0; ni < 2; ++ni) {
          const bf16_t* bp = bs0 + (wn + ni * 16 + lrow) * BKP + 8 * lhalf;
          v8bf blo = *(const v8bf*)(bp);
          v8bf bhi = *(const v8bf*)(bp + 16);
          v16bf bf;
#pragma unroll
          for (int e = 0; e < 8; ++e) { bf[e] = blo[e]; bf[e + 8] = bhi[e]; }
          acc[mi][ni] = __builtin_amdgcn_wmma_f32_16x16x32_bf16(
              false, af, false, bf, (short)0, acc[mi][ni], false, false);
        }
      }
    }
    if (more && bmode == 0) bf32_store(nsel);  // commit after math
    wait_asynccnt0();
    __syncthreads();
  }

  // ---- epilogue ----
  float* Cf = (float*)Cp + (long long)z * sC;
  bf16_t* Cb = (bf16_t*)Cp + (long long)z * sC;
#pragma unroll
  for (int mi = 0; mi < 2; ++mi)
#pragma unroll
    for (int ni = 0; ni < 2; ++ni)
#pragma unroll
      for (int r = 0; r < 8; ++r) {
        int mm = m0 + wm + mi * 16 + r + 8 * lhalf;
        int nn = n0 + wn + ni * 16 + lrow;
        float v = acc[mi][ni][r];
        if (cmode == 0)      Cf[(size_t)mm * ldc + nn] = v;
        else if (cmode == 1) Cb[(size_t)mm * ldc + nn] = f2bf(v);
        else                 Cb[(size_t)nn * ldc + mm] = f2bf(v);
      }
}

// ===================== elementwise / norm kernels =====================

__global__ void k_embed(const int* __restrict__ ids, const float* __restrict__ emb,
                        float* __restrict__ hidden) {
  int s = blockIdx.x;
  int id = ids[s];
  for (int c = threadIdx.x; c < HID; c += blockDim.x)
    hidden[(size_t)s * HID + c] = emb[(size_t)id * HID + c];
}

// y(bf16) = rms(x, w): feeds the next GEMM's A operand
__global__ void k_rmsnorm_bf(const float* __restrict__ x, const float* __restrict__ w,
                             bf16_t* __restrict__ y, int C) {
  int row = blockIdx.x;
  const float* xr = x + (size_t)row * C;
  float ss = 0.f;
  for (int c = threadIdx.x; c < C; c += blockDim.x) { float v = xr[c]; ss += v * v; }
  ss = blockReduceSum<256>(ss);
  float inv = rsqrtf(ss / C + 1e-6f);
  bf16_t* yr = y + (size_t)row * C;
  for (int c = threadIdx.x; c < C; c += blockDim.x)
    yr[c] = f2bf(xr[c] * inv * (1.f + w[c]));
}

// hidden(f32) += rms(src, w)  -- residual stream stays fp32
__global__ void k_rmsnorm_add(float* __restrict__ hidden, const float* __restrict__ src,
                              const float* __restrict__ w, int C) {
  int row = blockIdx.x;
  const float* xr = src + (size_t)row * C;
  float ss = 0.f;
  for (int c = threadIdx.x; c < C; c += blockDim.x) { float v = xr[c]; ss += v * v; }
  ss = blockReduceSum<256>(ss);
  float inv = rsqrtf(ss / C + 1e-6f);
  float* hr = hidden + (size_t)row * C;
  for (int c = threadIdx.x; c < C; c += blockDim.x)
    hr[c] += xr[c] * inv * (1.f + w[c]);
}

// fused per-head RMSNorm (qn/kn) + RoPE, in place on bf16 [S, nheads*HDIM]
__global__ void k_qknorm_rope(bf16_t* __restrict__ buf, const float* __restrict__ w,
                              const float* __restrict__ cosb, const float* __restrict__ sinb,
                              int nheads) {
  __shared__ float sh[HDIM];
  int h = blockIdx.x % nheads;
  int s = blockIdx.x / nheads;
  int d = threadIdx.x;                          // block = HDIM threads
  bf16_t* p = buf + ((size_t)s * nheads + h) * HDIM;
  float v = (float)p[d];
  float ss = blockReduceSum<HDIM>(v * v);
  float inv = rsqrtf(ss / HDIM + 1e-6f);
  float y = v * inv * (1.f + w[d]);
  sh[d] = y;
  __syncthreads();
  float rot = (d < HDIM / 2) ? -sh[d + HDIM / 2] : sh[d - HDIM / 2];
  float cv = cosb[(size_t)s * HDIM + d];        // cache_position[s] == s
  float sv = sinb[(size_t)s * HDIM + d];
  p[d] = f2bf(y * cv + rot * sv);
}

// causal softmax: fp32 scores in, bf16 probabilities out
__global__ void k_softmax(const float* __restrict__ sc, bf16_t* __restrict__ pr) {
  int s = blockIdx.x % S_LEN;
  int h = blockIdx.x / S_LEN;
  const float* row = sc + ((size_t)h * S_LEN + s) * S_LEN;
  bf16_t* orow = pr + ((size_t)h * S_LEN + s) * S_LEN;
  float mx = -1e30f;
#pragma unroll
  for (int i = 0; i < 2; ++i) {
    int t = threadIdx.x + i * 256;
    if (t <= s) mx = fmaxf(mx, row[t] * ATT_SCALE);
  }
  mx = blockReduceMax<256>(mx);
  float pv[2];
  float sum = 0.f;
#pragma unroll
  for (int i = 0; i < 2; ++i) {
    int t = threadIdx.x + i * 256;
    pv[i] = (t <= s) ? __expf(row[t] * ATT_SCALE - mx) : 0.f;
    sum += pv[i];
  }
  sum = blockReduceSum<256>(sum);
  float inv = 1.f / sum;
#pragma unroll
  for (int i = 0; i < 2; ++i) {
    int t = threadIdx.x + i * 256;
    orow[t] = f2bf(pv[i] * inv);
  }
}

// h2(bf16) = gelu_tanh(gate) * up
__global__ void k_gelu_mul(const float* __restrict__ g, const float* __restrict__ u,
                           bf16_t* __restrict__ o, int n) {
  int i = blockIdx.x * blockDim.x + threadIdx.x;
  if (i < n) {
    float x = g[i];
    float t = 0.7978845608028654f * (x + 0.044715f * x * x * x);
    o[i] = f2bf(0.5f * x * (1.f + tanhf(t)) * u[i]);
  }
}

// ============================ host driver ============================

extern "C" void kernel_launch(void* const* d_in, const int* in_sizes, int n_in,
                              void* d_out, int out_size, void* d_ws, size_t ws_size,
                              hipStream_t stream) {
  (void)in_sizes; (void)n_in; (void)out_size; (void)ws_size;
  const int*   ids  = (const int*)  d_in[0];
  // d_in[1] = cache_position (== arange(S), folded into the kernels)
  const float* emb  = (const float*)d_in[2];
  const float* Wq   = (const float*)d_in[3];
  const float* Wk   = (const float*)d_in[4];
  const float* Wv   = (const float*)d_in[5];
  const float* Wo   = (const float*)d_in[6];
  const float* qn   = (const float*)d_in[7];
  const float* kn   = (const float*)d_in[8];
  const float* ln1  = (const float*)d_in[9];
  const float* ln2  = (const float*)d_in[10];
  const float* ln3  = (const float*)d_in[11];
  const float* ln4  = (const float*)d_in[12];
  const float* Wg   = (const float*)d_in[13];
  const float* Wu   = (const float*)d_in[14];
  const float* Wd   = (const float*)d_in[15];
  const float* fn   = (const float*)d_in[16];
  const float* lm   = (const float*)d_in[17];
  const float* rc   = (const float*)d_in[18];
  const float* rs   = (const float*)d_in[19];
  float* out = (float*)d_out;

  // ---- workspace carve: fp32 region, then bf16 region ----
  float* wsf = (float*)d_ws;
  size_t off = 0;
  float* hidden = wsf + off; off += (size_t)S_LEN * HID;          // 1M
  float* scores = wsf + off; off += (size_t)NHEAD * S_LEN * S_LEN;// 2M
  float* gate   = wsf + off; off += (size_t)S_LEN * INTER_DIM;    // 4M
  float* up     = wsf + off; off += (size_t)S_LEN * INTER_DIM;    // 4M
  bf16_t* wsb = (bf16_t*)(wsf + off);
  size_t ob = 0;
  bf16_t* xbf   = wsb + ob; ob += (size_t)S_LEN * HID;
  bf16_t* qb    = wsb + ob; ob += (size_t)S_LEN * NHEAD * HDIM;
  bf16_t* kb    = wsb + ob; ob += (size_t)S_LEN * KVHEAD * HDIM;
  bf16_t* vT    = wsb + ob; ob += (size_t)KVHEAD * HDIM * S_LEN;  // [kh][d][s]
  bf16_t* probs = wsb + ob; ob += (size_t)NHEAD * S_LEN * S_LEN;
  bf16_t* attno = wsb + ob; ob += (size_t)S_LEN * NHEAD * HDIM;
  bf16_t* h2    = wsb + ob; ob += (size_t)S_LEN * INTER_DIM;

  auto gemm = [&](const bf16_t* A, const void* B, void* C, int M, int N, int K,
                  int lda, int ldb, int ldc, int batch,
                  long long sA, int dA, long long sB, int dB, long long sC,
                  int bmode, int cmode) {
    dim3 grid(N / BN, M / BM, batch);
    k_gemm<<<grid, GT, 0, stream>>>(A, B, C, M, N, K, lda, ldb, ldc,
                                    sA, dA, sB, dB, sC, bmode, cmode);
  };

  k_embed<<<S_LEN, 256, 0, stream>>>(ids, emb, hidden);

  for (int i = 0; i < NLAYER; ++i) {
    const float* Wq_i = Wq + (size_t)i * HID * NHEAD * HDIM;
    const float* Wk_i = Wk + (size_t)i * HID * KVHEAD * HDIM;
    const float* Wv_i = Wv + (size_t)i * HID * KVHEAD * HDIM;
    const float* Wo_i = Wo + (size_t)i * NHEAD * HDIM * HID;
    const float* Wg_i = Wg + (size_t)i * HID * INTER_DIM;
    const float* Wu_i = Wu + (size_t)i * HID * INTER_DIM;
    const float* Wd_i = Wd + (size_t)i * INTER_DIM * HID;

    // x = rms(hidden, ln1) -> bf16
    k_rmsnorm_bf<<<S_LEN, 256, 0, stream>>>(hidden, ln1 + (size_t)i * HID, xbf, HID);
    // q, k projections -> bf16
    gemm(xbf, Wq_i, qb, S_LEN, NHEAD * HDIM, HID, HID, NHEAD * HDIM, NHEAD * HDIM,
         1, 0, 1, 0, 1, 0, /*bmode*/0, /*cmode*/1);
    gemm(xbf, Wk_i, kb, S_LEN, KVHEAD * HDIM, HID, HID, KVHEAD * HDIM, KVHEAD * HDIM,
         1, 0, 1, 0, 1, 0, 0, 1);
    // v projection, per-kv-head batched, stored transposed: vT[kh][d][s]
    gemm(xbf, Wv_i, vT, S_LEN, HDIM, HID, HID, KVHEAD * HDIM, S_LEN,
         KVHEAD, 0, 1, HDIM, 1, (long long)HDIM * S_LEN, 0, /*cmode*/2);
    // qk-norm + RoPE (in place, bf16)
    k_qknorm_rope<<<S_LEN * NHEAD, HDIM, 0, stream>>>(qb, qn + (size_t)i * HDIM, rc, rs, NHEAD);
    k_qknorm_rope<<<S_LEN * KVHEAD, HDIM, 0, stream>>>(kb, kn + (size_t)i * HDIM, rc, rs, KVHEAD);
    // scores[h] = q[h] @ k[h/2]^T  (bf16 async B, fp32 C for softmax)
    gemm(qb, kb, scores, S_LEN, S_LEN, HDIM,
         NHEAD * HDIM, KVHEAD * HDIM, S_LEN, NHEAD,
         HDIM, 1, HDIM, 2, (long long)S_LEN * S_LEN, /*bmode*/1, /*cmode*/0);
    k_softmax<<<NHEAD * S_LEN, 256, 0, stream>>>(scores, probs);
    // attno[:, h*HD:(h+1)*HD] = probs[h] @ v[h/2]  (B = vT, bf16 [N,K])
    gemm(probs, vT, attno, S_LEN, HDIM, S_LEN,
         S_LEN, S_LEN, NHEAD * HDIM, NHEAD,
         (long long)S_LEN * S_LEN, 1, (long long)HDIM * S_LEN, 2, HDIM, 1, 1);
    // o = attno @ Wo -> fp32 scratch (gate)
    gemm(attno, Wo_i, gate, S_LEN, HID, NHEAD * HDIM, NHEAD * HDIM, HID, HID,
         1, 0, 1, 0, 1, 0, 0, 0);
    k_rmsnorm_add<<<S_LEN, 256, 0, stream>>>(hidden, gate, ln2 + (size_t)i * HID, HID);
    // MLP
    k_rmsnorm_bf<<<S_LEN, 256, 0, stream>>>(hidden, ln3 + (size_t)i * HID, xbf, HID);
    gemm(xbf, Wg_i, gate, S_LEN, INTER_DIM, HID, HID, INTER_DIM, INTER_DIM,
         1, 0, 1, 0, 1, 0, 0, 0);
    gemm(xbf, Wu_i, up, S_LEN, INTER_DIM, HID, HID, INTER_DIM, INTER_DIM,
         1, 0, 1, 0, 1, 0, 0, 0);
    int nmm = S_LEN * INTER_DIM;
    k_gelu_mul<<<(nmm + 255) / 256, 256, 0, stream>>>(gate, up, h2, nmm);
    gemm(h2, Wd_i, gate, S_LEN, HID, INTER_DIM, INTER_DIM, HID, HID,
         1, 0, 1, 0, 1, 0, 0, 0);
    k_rmsnorm_add<<<S_LEN, 256, 0, stream>>>(hidden, gate, ln4 + (size_t)i * HID, HID);
  }

  // final norm + LM head
  k_rmsnorm_bf<<<S_LEN, 256, 0, stream>>>(hidden, fn, xbf, HID);
  gemm(xbf, lm, out, S_LEN, VOCAB, HID, HID, VOCAB, VOCAB, 1, 0, 1, 0, 1, 0, 0, 0);
}
